// RNN_23192823398477
// MI455X (gfx1250) — matrix operand order repaired and checked by reading, not compiled, not run on hardware
//
#include <hip/hip_runtime.h>

// Persistent fused RNN for MI455X (gfx1250, wave32, WMMA).
//   grid = 16 WGs (16 batch rows each), block = 256 threads = 8 waves.
//   wave w: recurrence N-tile w (4x v_wmma_f32_16x16x32_f16 per step, 2x2 chains)
//   waves 0-3: + FC1 tile w  (1 step behind, reuses the A registers)
//   waves 4-6: FC2 output component (2 steps behind, VALU dots)
//   wave 7:  + x prefetch into LDS
// All state lives in LDS/VGPRs; HBM traffic ~13 MB total (x in, out + h_T out).

#define B_TOT  256
#define T_LEN  2048
#define HID    128
#define NFC1   64
#define NOUT   3
#define BT     16     // batch tile per workgroup
#define HPITCH 136    // halves per hbuf row (128 + 8 pad -> 272B, bank-skewed)
#define FPITCH 68     // floats per fc1buf row (64 + 4 pad)

typedef _Float16 v8h  __attribute__((ext_vector_type(8)));
typedef _Float16 v16h __attribute__((ext_vector_type(16)));
typedef float    v8f  __attribute__((ext_vector_type(8)));

template <bool V> struct Flag { static constexpr bool value = V; };

__device__ __forceinline__ v16h cat8(v8h lo, v8h hi) {
  return __builtin_shufflevector(lo, hi, 0,1,2,3,4,5,6,7,8,9,10,11,12,13,14,15);
}

__device__ __forceinline__ float fast_tanh(float x) {
#if __has_builtin(__builtin_amdgcn_tanhf)
  return __builtin_amdgcn_tanhf(x);          // gfx1250 V_TANH_F32 (trans op)
#else
  float xc = fminf(fmaxf(x, -9.f), 9.f);
  float e  = __builtin_amdgcn_exp2f(xc * 2.8853900817779268f);
  return (e - 1.f) * __builtin_amdgcn_rcpf(e + 1.f);
#endif
}

#define WMMA(A, Bm, C) \
  __builtin_amdgcn_wmma_f32_16x16x32_f16(false, (A), false, (Bm), (short)0, (C), false, false)

__global__ __launch_bounds__(256, 1)
void rnn_persistent(const float* __restrict__ x,
                    const float* __restrict__ hidden0,
                    const float* __restrict__ W_ih,
                    const float* __restrict__ W_hh,
                    const float* __restrict__ b_ih,
                    const float* __restrict__ b_hh,
                    const float* __restrict__ W_fc1,
                    const float* __restrict__ b_fc1,
                    const float* __restrict__ W_fc2,
                    const float* __restrict__ b_fc2,
                    float* __restrict__ out)
{
  __shared__ __align__(16) _Float16 hbuf[2][BT * HPITCH];  // h state, f16, ping-pong
  __shared__ __align__(16) float    fc1buf[2][BT * FPITCH];
  __shared__ __align__(16) float    xbuf[2][BT * 4];       // x tile staging
  __shared__ __align__(16) float    wfc2s[NOUT * NFC1];

  const int tid   = threadIdx.x;
  const int wid   = tid >> 5;
  const int lane  = tid & 31;
  const int nloc  = lane & 15;
  const int hi16  = lane >> 4;        // lane half (0/1)
  const int mbase = hi16 * 8;         // C/D rows owned by this lane
  const int kh    = hi16 * 8;         // A-matrix K sub-offset for this lane half
  const int ncol  = wid * 16 + nloc;  // hidden column (rec) / fc1 feature index
  const int b0    = blockIdx.x * BT;

  // ---------------- one-time preloads ----------------
  v16h bhh[4];                         // W_hh B-tiles: B[k,n] = W_hh[n,k]
#pragma unroll
  for (int kt = 0; kt < 4; ++kt) {
    const float* src = W_hh + (size_t)ncol * HID + kt * 32 + hi16 * 8;
#pragma unroll
    for (int j = 0; j < 16; ++j) bhh[kt][j] = (_Float16)src[j < 8 ? j : j + 8];
  }
  const float wih0 = W_ih[ncol * 3 + 0];
  const float wih1 = W_ih[ncol * 3 + 1];
  const float wih2 = W_ih[ncol * 3 + 2];
  const float biasv = b_ih[ncol] + b_hh[ncol];

  v16h  bf1[4] = {};
  float bfc1v  = 0.f;
  if (wid < 4) {                       // FC1 weight tiles (64 outputs -> waves 0..3)
#pragma unroll
    for (int kt = 0; kt < 4; ++kt) {
      const float* src = W_fc1 + (size_t)ncol * HID + kt * 32 + hi16 * 8;
#pragma unroll
      for (int j = 0; j < 16; ++j) bf1[kt][j] = (_Float16)src[j < 8 ? j : j + 8];
    }
    bfc1v = b_fc1[ncol];
  }
  const float bfc2v = (wid >= 4 && wid < 7) ? b_fc2[wid - 4] : 0.f;

  if (tid < NOUT * NFC1) wfc2s[tid] = W_fc2[tid];

  // H_0 -> hbuf[0] (f16), x(t=0) -> xbuf[0]
  {
    const int r = tid >> 4, c0 = (tid & 15) * 8;
#pragma unroll
    for (int j = 0; j < 8; ++j)
      hbuf[0][r * HPITCH + c0 + j] =
          (_Float16)hidden0[(size_t)(b0 + r) * HID + c0 + j];
  }
  if (tid < BT) {
    const float* xs = x + (size_t)(b0 + tid) * T_LEN * 3;
    xbuf[0][tid * 4 + 0] = xs[0];
    xbuf[0][tid * 4 + 1] = xs[1];
    xbuf[0][tid * 4 + 2] = xs[2];
    xbuf[0][tid * 4 + 3] = 0.f;
  }

  // ---------------- one pipeline step (phase flags are compile-time) --------
  auto step = [&](int i, auto REC, auto FC1F, auto FC2F, auto XPRE, auto HT) {
    const int p = i & 1;
    __syncthreads();

    // wave 7: prefetch x(i+1) — issue loads first, publish to LDS at the end
    float xf0 = 0.f, xf1 = 0.f, xf2 = 0.f;
    if constexpr (XPRE.value) {
      if (wid == 7 && lane < BT) {
        const float* xs = x + ((size_t)(b0 + lane) * T_LEN + (i + 1)) * 3;
        xf0 = xs[0]; xf1 = xs[1]; xf2 = xs[2];
      }
    }

    // A-matrix = h_i from hbuf[p] (shared by recurrence and FC1)
    v16h a0 = {}, a1 = {}, a2 = {}, a3 = {};
    if constexpr (REC.value || FC1F.value) {
      if (REC.value || wid < 4) {
        const _Float16* hb = hbuf[p] + nloc * HPITCH;   // A row = lane&15
        a0 = cat8(*(const v8h*)(hb + 0  + kh), *(const v8h*)(hb + 0  + kh + 16));
        a1 = cat8(*(const v8h*)(hb + 32 + kh), *(const v8h*)(hb + 32 + kh + 16));
        a2 = cat8(*(const v8h*)(hb + 64 + kh), *(const v8h*)(hb + 64 + kh + 16));
        a3 = cat8(*(const v8h*)(hb + 96 + kh), *(const v8h*)(hb + 96 + kh + 16));
      }
    }

    // recurrence: h_{i+1} = tanh(xp_i + h_i @ W_hh^T)   (all 8 waves)
    if constexpr (REC.value) {
      const v8f z = {};
      v8f acc  = WMMA(a0, bhh[0], z);    // two independent 2-deep WMMA chains,
      v8f acc2 = WMMA(a2, bhh[2], z);    // C=0 so WMMA issue isn't gated on xp
      acc  = WMMA(a1, bhh[1], acc);
      acc2 = WMMA(a3, bhh[3], acc2);
      v8f xp;                            // input projection co-executes with WMMA
#pragma unroll
      for (int j = 0; j < 8; ++j) {
        const float* xr = &xbuf[p][(mbase + j) * 4];
        xp[j] = xr[0] * wih0 + xr[1] * wih1 + xr[2] * wih2 + biasv;
      }
      _Float16* hn = hbuf[1 - p];
#pragma unroll
      for (int j = 0; j < 8; ++j) {
        float hv = fast_tanh(acc[j] + (acc2[j] + xp[j]));
        hn[(mbase + j) * HPITCH + ncol] = (_Float16)hv;
        if constexpr (HT.value)          // final hidden -> second output tensor
          out[(size_t)B_TOT * T_LEN * 3 + (size_t)(b0 + mbase + j) * HID + ncol] = hv;
      }
    }

    // FC1 on h_i (output row t = i-1), waves 0..3, same A registers
    if constexpr (FC1F.value) {
      if (wid < 4) {
        v8f f, f2 = {};
#pragma unroll
        for (int j = 0; j < 8; ++j) f[j] = bfc1v;
        f  = WMMA(a0, bf1[0], f);
        f2 = WMMA(a2, bf1[2], f2);
        f  = WMMA(a1, bf1[1], f);
        f2 = WMMA(a3, bf1[3], f2);
        float* fb = fc1buf[p];
#pragma unroll
        for (int j = 0; j < 8; ++j)
          fb[(mbase + j) * FPITCH + ncol] = f[j] + f2[j];
      }
    }

    // FC2 + store (output row t = i-2), waves 4..6, lane = batch row
    if constexpr (FC2F.value) {
      if (wid >= 4 && wid < 7 && lane < BT) {
        const int o = wid - 4, m = lane, trow = i - 2;
        const float* fr = &fc1buf[1 - p][m * FPITCH];
        float acc = bfc2v;
#pragma unroll
        for (int fidx = 0; fidx < NFC1; fidx += 4) {
          const float* a = fr + fidx;
          const float* w = &wfc2s[o * NFC1 + fidx];
          acc += a[0] * w[0] + a[1] * w[1] + a[2] * w[2] + a[3] * w[3];
        }
        out[((size_t)(b0 + m) * T_LEN + trow) * 3 + o] = acc;
      }
    }

    if constexpr (XPRE.value) {          // publish next x tile
      if (wid == 7 && lane < BT) {
        float* xd = &xbuf[1 - p][lane * 4];
        xd[0] = xf0; xd[1] = xf1; xd[2] = xf2; xd[3] = 0.f;
      }
    }
  };

  // -------- phase schedule: prologue / branch-free steady state / drain ----
  step(0, Flag<true>{},  Flag<false>{}, Flag<false>{}, Flag<true>{},  Flag<false>{});
  step(1, Flag<true>{},  Flag<true>{},  Flag<false>{}, Flag<true>{},  Flag<false>{});
  for (int i = 2; i < T_LEN - 1; ++i)
    step(i, Flag<true>{}, Flag<true>{}, Flag<true>{},  Flag<true>{},  Flag<false>{});
  step(T_LEN - 1, Flag<true>{},  Flag<true>{},  Flag<true>{}, Flag<false>{}, Flag<true>{});
  step(T_LEN,     Flag<false>{}, Flag<true>{},  Flag<true>{}, Flag<false>{}, Flag<false>{});
  step(T_LEN + 1, Flag<false>{}, Flag<false>{}, Flag<true>{}, Flag<false>{}, Flag<false>{});
}

extern "C" void kernel_launch(void* const* d_in, const int* in_sizes, int n_in,
                              void* d_out, int out_size, void* d_ws, size_t ws_size,
                              hipStream_t stream) {
  (void)in_sizes; (void)n_in; (void)out_size; (void)d_ws; (void)ws_size;
  const float* x     = (const float*)d_in[0];
  const float* h0    = (const float*)d_in[1];
  const float* W_ih  = (const float*)d_in[2];
  const float* W_hh  = (const float*)d_in[3];
  const float* b_ih  = (const float*)d_in[4];
  const float* b_hh  = (const float*)d_in[5];
  const float* W_fc1 = (const float*)d_in[6];
  const float* b_fc1 = (const float*)d_in[7];
  const float* W_fc2 = (const float*)d_in[8];
  const float* b_fc2 = (const float*)d_in[9];
  float* out = (float*)d_out;

  rnn_persistent<<<B_TOT / BT, 256, 0, stream>>>(
      x, h0, W_ih, W_hh, b_ih, b_hh, W_fc1, b_fc1, W_fc2, b_fc2, out);
}
